// ModelRichContext_54013508715187
// MI455X (gfx1250) — matrix operand order repaired and checked by reading, not compiled, not run on hardware
//
#include <hip/hip_runtime.h>
#include <math.h>
#include <stdint.h>

// N=16, L=256, S=128, E=128, A=9, D=768, N_HID=768, ARG_ROLES=23
typedef __bf16 bf16;
typedef __attribute__((ext_vector_type(16))) __bf16 v16bf;
typedef __attribute__((ext_vector_type(8)))  __bf16 v8bf;
typedef __attribute__((ext_vector_type(8)))  float  v8f;

// ---------------------------------------------------------------------------
// CDNA5 async global->LDS copy (ISA 15.18.3 op 98) + ASYNCcnt waits.
// LDS destination address = low 32 bits of the generic shared pointer
// (shared aperture is {SHARED_BASE, offset[31:0]}).
// ---------------------------------------------------------------------------
__device__ __forceinline__ void async_ld16(unsigned ldsOff, const void* g) {
    asm volatile("global_load_async_to_lds_b128 %0, %1, off"
                 :: "v"(ldsOff), "v"(g) : "memory");
}
__device__ __forceinline__ void wait_async0() {
    asm volatile("s_wait_asynccnt 0x0" ::: "memory");
}
__device__ __forceinline__ void wait_async6() {   // keep next tile's 6 copies in flight
    asm volatile("s_wait_asynccnt 0x6" ::: "memory");
}

// WMMA fragment builders (ISA 7.12.2 16-bit layouts), from padded LDS tiles
// (row stride 72 bf16 = 144 B keeps every 16B read aligned).
__device__ __forceinline__ v16bf frag_a(const bf16* base, int row, int kofs) {
    v8bf lo = *(const v8bf*)(base + row * 72 + kofs);        // K kofs..+7
    v8bf hi = *(const v8bf*)(base + row * 72 + kofs + 16);   // K kofs+16..+23
    return __builtin_shufflevector(lo, hi, 0,1,2,3,4,5,6,7,8,9,10,11,12,13,14,15);
}
__device__ __forceinline__ v16bf frag_b(const bf16* base, int nrow, int kofs) {
    v8bf lo = *(const v8bf*)(base + nrow * 72 + kofs);       // K kofs..+7
    v8bf hi = *(const v8bf*)(base + nrow * 72 + kofs + 16);  // K kofs+16..+23
    return __builtin_shufflevector(lo, hi, 0,1,2,3,4,5,6,7,8,9,10,11,12,13,14,15);
}

// ---------------------------------------------------------------------------
// K0: reorder W1 row-blocks, convert to bf16, store TRANSPOSED (n-major):
// W1entT[n][k] = W1[{0,2,3}[k/768]*768 + k%768][n]   ({1,4,5} for arg).
// ---------------------------------------------------------------------------
__global__ void prep_w1_kernel(const float* __restrict__ W1,
                               bf16* __restrict__ W1entT, bf16* __restrict__ W1argT) {
    int idx = blockIdx.x * 256 + threadIdx.x;
    const int per = 768 * 2304;
    if (idx >= 2 * per) return;
    int which = idx / per;
    int rem   = idx % per;
    int n = rem / 2304, k = rem % 2304;
    int b = k / 768, sub = k % 768;
    const int entm[3] = {0, 2, 3}, argm[3] = {1, 4, 5};
    int src = (which ? argm[b] : entm[b]) * 768 + sub;
    bf16 v = (bf16)W1[(size_t)src * 768 + n];
    (which ? W1argT : W1entT)[(size_t)n * 2304 + k] = v;
}

// ---------------------------------------------------------------------------
// K1: per-batch gathers: sent_emb transposed bf16 (D x S, K-contiguous B
// operand), entity_mapping^T bf16 (A operand), arg_emb f32 (K=9 VALU).
// ---------------------------------------------------------------------------
__global__ __launch_bounds__(256)
void gather_kernel(const float* __restrict__ allemb, const float* __restrict__ entmap,
                   const float* __restrict__ argW, const int* __restrict__ idxs,
                   bf16* __restrict__ sentBT, bf16* __restrict__ emTB,
                   float* __restrict__ argembF) {
    __shared__ float aw[81];
    int n = blockIdx.x, tid = threadIdx.x;
    if (tid < 81) aw[tid] = argW[n * 81 + tid];
    __syncthreads();
    const float* embn = allemb + (size_t)n * 256 * 768;
    for (int i = tid; i < 128 * 768; i += 256) {
        int s = i / 768, d = i % 768;
        int r = idxs[n * 130 + s];
        float v = embn[(size_t)r * 768 + d];
        sentBT[(size_t)n * 768 * 128 + (size_t)d * 128 + s] = (bf16)v;
    }
    for (int i = tid; i < 128 * 128; i += 256) {
        int e = i / 128, s = i % 128;
        emTB[(size_t)n * 128 * 128 + i] = (bf16)entmap[((size_t)n * 128 + s) * 128 + e];
    }
    int base = idxs[n * 130 + 128] + 1;
    for (int i = tid; i < 9 * 768; i += 256) {
        int a = i / 768, d = i % 768;
        float t = 0.f;
        #pragma unroll
        for (int k = 0; k < 9; k++)
            t += embn[(size_t)(base + k) * 768 + d] * aw[k * 9 + a];
        argembF[(size_t)n * 9 * 768 + i] = t;
    }
}

// ---------------------------------------------------------------------------
// K2: t2t(n,i,j) = mean_l [row-normalized head-mean gathered attention].
// Reads only the gathered 3x12x128x128 slice (37.7 MB of the 151 MB tensor).
// ---------------------------------------------------------------------------
__global__ __launch_bounds__(128)
void t2t_kernel(const float* __restrict__ attn, const int* __restrict__ idxs,
                bf16* __restrict__ t2tB) {
    __shared__ float red[128];
    int blk = blockIdx.x;
    int n = blk >> 7, i = blk & 127, tid = threadIdx.x;
    int row = idxs[n * 130 + i];
    int col = idxs[n * 130 + tid];
    float acc = 0.f;
    for (int l = 0; l < 3; l++) {
        size_t base = ((size_t)(l * 16 + n)) * 12 * 256 * 256;
        float v = 0.f;
        #pragma unroll
        for (int h = 0; h < 12; h++)
            v += attn[base + ((size_t)h * 256 + row) * 256 + col];
        v *= (1.0f / 12.0f);
        red[tid] = v;
        __syncthreads();
        for (int o = 64; o > 0; o >>= 1) {
            if (tid < o) red[tid] += red[tid + o];
            __syncthreads();
        }
        float rs = red[0];
        __syncthreads();
        acc += v / (rs + 1e-9f);
    }
    t2tB[((size_t)(n * 128 + i)) * 128 + tid] = (bf16)(acc * (1.0f / 3.0f));
}

// ---------------------------------------------------------------------------
// K3: batched bf16 WMMA GEMM  C[M x Nc] = A[M x K] @ BT[Nc x K]^T (f32 acc).
// 256 threads = 8 waves (2x4 wave grid); each wave computes a 2x2 quad of
// 16x16 tiles -> 64x128 block tile, 8 WMMAs per 64-wide K chunk.
// Tiles staged via global_load_async_to_lds_b128 into DOUBLE-BUFFERED LDS:
// next chunk's 6 copies issued before computing, s_wait_asynccnt 6 keeps
// them in flight behind the WMMAs.
// flags: 1 -> f32 C ; 2 -> bf16 C row-major ; 4 -> bf16 C transposed (ldcb).
// ---------------------------------------------------------------------------
__global__ __launch_bounds__(256)
void wmma_gemm_bf16(const bf16* __restrict__ A, const bf16* __restrict__ BT,
                    float* __restrict__ Cf, bf16* __restrict__ Cb,
                    int K, int lda, int ldbt, int ldc, int ldcb,
                    long strideA, long strideB, long strideC, int flags) {
    __shared__ __align__(16) bf16 ldsA[2][64 * 72];
    __shared__ __align__(16) bf16 ldsB[2][128 * 72];
    const int bz = blockIdx.z;
    A  += (size_t)bz * strideA;
    BT += (size_t)bz * strideB;
    const int tM = blockIdx.y * 64, tN = blockIdx.x * 128;
    const int tid = threadIdx.x;
    const int wave = tid >> 5, lane = tid & 31;
    const int waveM = (wave >> 2) * 32;   // 0 / 32
    const int waveN = (wave & 3) * 32;    // 0..96

    // staging: 16B chunk per copy; A rows rS, rS+32; B rows rS..rS+96
    const int rS = tid >> 3;              // 0..31
    const int cS = (tid & 7) * 8;         // bf16 elems 0..56
    const bf16* gA0 = A  + (size_t)(tM + rS)      * lda  + cS;
    const bf16* gA1 = A  + (size_t)(tM + rS + 32) * lda  + cS;
    const bf16* gB0 = BT + (size_t)(tN + rS)      * ldbt + cS;
    const bf16* gB1 = BT + (size_t)(tN + rS + 32) * ldbt + cS;
    const bf16* gB2 = BT + (size_t)(tN + rS + 64) * ldbt + cS;
    const bf16* gB3 = BT + (size_t)(tN + rS + 96) * ldbt + cS;
    const unsigned lA0 = (unsigned)(uintptr_t)(&ldsA[0][rS * 72 + cS]);
    const unsigned lB0 = (unsigned)(uintptr_t)(&ldsB[0][rS * 72 + cS]);
    const unsigned bufA = (unsigned)(64 * 72 * 2);   // bytes between buffers
    const unsigned bufB = (unsigned)(128 * 72 * 2);
    const unsigned rowA = (unsigned)(32 * 72 * 2);

#define STAGE(p, k0)                                                        \
    do {                                                                    \
        unsigned ao = (p) ? bufA : 0u, bo = (p) ? bufB : 0u;                \
        async_ld16(lA0 + ao,              (const void*)(gA0 + (k0)));       \
        async_ld16(lA0 + ao + rowA,       (const void*)(gA1 + (k0)));       \
        async_ld16(lB0 + bo,              (const void*)(gB0 + (k0)));       \
        async_ld16(lB0 + bo + rowA,       (const void*)(gB1 + (k0)));       \
        async_ld16(lB0 + bo + 2 * rowA,   (const void*)(gB2 + (k0)));       \
        async_ld16(lB0 + bo + 3 * rowA,   (const void*)(gB3 + (k0)));       \
    } while (0)

    v8f c[2][2];
    #pragma unroll
    for (int i = 0; i < 2; i++)
        #pragma unroll
        for (int j = 0; j < 2; j++)
            #pragma unroll
            for (int r = 0; r < 8; r++) c[i][j][r] = 0.0f;

    const int m0  = (lane & 15) + waveM;   // A fragment rows
    const int n0  = (lane & 15) + waveN;   // B fragment columns
    const int kb  = (lane >> 4) * 8;       // A: K kb..+7 & kb+16..+23
    const int kb2 = (lane >> 4) * 8;       // B: K kb2..+7 & kb2+16..+23

    STAGE(0, 0);
    for (int k0 = 0; k0 < K; k0 += 64) {
        const int p = (k0 >> 6) & 1;
        if (k0 + 64 < K) {
            STAGE(p ^ 1, k0 + 64);
            __builtin_prefetch((const void*)(gB0 + k0 + 128), 0, 1);
            wait_async6();
        } else {
            wait_async0();
        }
        __syncthreads();
        const bf16* Ab = ldsA[p];
        const bf16* Bb = ldsB[p];
        #pragma unroll
        for (int kk = 0; kk < 64; kk += 32) {
            v16bf a0 = frag_a(Ab, m0,      kk + kb);
            v16bf a1 = frag_a(Ab, m0 + 16, kk + kb);
            v16bf b0 = frag_b(Bb, n0,      kk + kb2);
            v16bf b1 = frag_b(Bb, n0 + 16, kk + kb2);
            c[0][0] = __builtin_amdgcn_wmma_f32_16x16x32_bf16(false, a0, false, b0, (short)0, c[0][0], false, false);
            c[0][1] = __builtin_amdgcn_wmma_f32_16x16x32_bf16(false, a0, false, b1, (short)0, c[0][1], false, false);
            c[1][0] = __builtin_amdgcn_wmma_f32_16x16x32_bf16(false, a1, false, b0, (short)0, c[1][0], false, false);
            c[1][1] = __builtin_amdgcn_wmma_f32_16x16x32_bf16(false, a1, false, b1, (short)0, c[1][1], false, false);
        }
        __syncthreads();
    }
#undef STAGE

    // C/D layout: VGPR r -> row r (+8 for upper half-lanes), col = lane&15
    #pragma unroll
    for (int ti = 0; ti < 2; ti++) {
        #pragma unroll
        for (int tj = 0; tj < 2; tj++) {
            int rowb = tM + waveM + ti * 16 + ((lane >> 4) * 8);
            int colg = tN + waveN + tj * 16 + (lane & 15);
            #pragma unroll
            for (int r = 0; r < 8; r++) {
                int rowg = rowb + r;
                if (flags & 1) Cf[(size_t)bz * strideC + (size_t)rowg * ldc + colg] = c[ti][tj][r];
                if (flags & 2) Cb[(size_t)bz * strideC + (size_t)rowg * ldc + colg] = (bf16)c[ti][tj][r];
                if (flags & 4) Cb[(size_t)bz * strideC + (size_t)colg * ldcb + rowg] = (bf16)c[ti][tj][r];
            }
        }
    }
}

// ---------------------------------------------------------------------------
// K4: per-batch VALU interactions (K=9 contractions too small for WMMA):
//  w=(t2a/sqrtD-5)/2, token_argAwared, arg_tokenAwared, a2a softmax, u2u;
//  assembles bf16 EntFeats (N*E x 2304) and ArgFeats (N*16 x 2304, padded).
// ---------------------------------------------------------------------------
__global__ __launch_bounds__(256)
void interactions_kernel(const float* __restrict__ entF, const float* __restrict__ argembF,
                         const bf16* __restrict__ ah2hB,
                         bf16* __restrict__ EntFeats, bf16* __restrict__ ArgFeats) {
    __shared__ float argE[9 * 768];
    __shared__ float wbuf[128 * 9];
    __shared__ float gmat[81];
    __shared__ float smax[81];
    int n = blockIdx.x, tid = threadIdx.x;
    const float rsqrtD = 0.036084391824352f;  // 1/sqrt(768)
    const float* entFn = entF + (size_t)n * 128 * 768;

    for (int i = tid; i < 9 * 768; i += 256) argE[i] = argembF[(size_t)n * 9 * 768 + i];
    __syncthreads();

    for (int i = tid; i < 128 * 9; i += 256) {
        int e = i / 9, a = i % 9;
        const float* er = entFn + (size_t)e * 768;
        const float* ar = argE + a * 768;
        float s = 0.f;
        for (int d = 0; d < 768; d++) s += er[d] * ar[d];
        wbuf[i] = (s * rsqrtD - 5.0f) * 0.5f;
    }
    __syncthreads();

    for (int i = tid; i < 128 * 768; i += 256) {
        int e = i / 768, d = i % 768;
        float t = 0.f;
        #pragma unroll
        for (int a = 0; a < 9; a++) t += wbuf[e * 9 + a] * argE[a * 768 + d];
        size_t base = ((size_t)(n * 128 + e)) * 2304;
        EntFeats[base + d]        = (bf16)entFn[(size_t)e * 768 + d];
        EntFeats[base + 768 + d]  = (bf16)t;
        EntFeats[base + 1536 + d] = ah2hB[(size_t)n * 128 * 768 + i];
    }

    for (int i = tid; i < 9 * 768; i += 256) {
        int a = i / 768, d = i % 768;
        float t = 0.f;
        for (int e = 0; e < 128; e++) t += wbuf[e * 9 + a] * entFn[(size_t)e * 768 + d];
        size_t base = ((size_t)(n * 16 + a)) * 2304;
        ArgFeats[base + d]       = (bf16)argE[i];
        ArgFeats[base + 768 + d] = (bf16)t;
    }

    if (tid < 81) {
        int a = tid / 9, b = tid % 9;
        float s = 0.f;
        for (int d = 0; d < 768; d++) s += argE[a * 768 + d] * argE[b * 768 + d];
        gmat[tid] = s;
    }
    __syncthreads();
    if (tid < 9) {
        float mx = -1e30f;
        for (int b = 0; b < 9; b++) mx = fmaxf(mx, gmat[tid * 9 + b]);
        float ssum = 0.f;
        for (int b = 0; b < 9; b++) { float ev = __expf(gmat[tid * 9 + b] - mx); smax[tid * 9 + b] = ev; ssum += ev; }
        for (int b = 0; b < 9; b++) smax[tid * 9 + b] /= ssum;
    }
    __syncthreads();

    for (int i = tid; i < 9 * 768; i += 256) {
        int a = i / 768, d = i % 768;
        float t = 0.f;
        #pragma unroll
        for (int b = 0; b < 9; b++) t += smax[a * 9 + b] * argE[b * 768 + d];
        ArgFeats[((size_t)(n * 16 + a)) * 2304 + 1536 + d] = (bf16)t;
    }
    for (int i = tid; i < 7 * 2304; i += 256) {
        int a = 9 + i / 2304, col = i % 2304;
        ArgFeats[((size_t)(n * 16 + a)) * 2304 + col] = (bf16)0.0f;
    }
}

// ---------------------------------------------------------------------------
// K5: score(n,e,a) = sum_h gelu(ent_part + arg_part + b1) * W2 + b2, then
// scatter into (E, 24) with -1e6 fill. One block per (n,e).
// ---------------------------------------------------------------------------
__global__ __launch_bounds__(256)
void score_kernel(const float* __restrict__ entPart, const float* __restrict__ argPart,
                  const float* __restrict__ b1, const float* __restrict__ W2,
                  const float* __restrict__ b2, const int* __restrict__ argmap,
                  float* __restrict__ out) {
    __shared__ float part[9][8];
    __shared__ float sc[9];
    int blk = blockIdx.x;
    int n = blk >> 7, tid = threadIdx.x;
    int wid = tid >> 5, lane = tid & 31;
    const float* ep = entPart + (size_t)blk * 768;
    float acc[9];
    #pragma unroll
    for (int a = 0; a < 9; a++) acc[a] = 0.f;
    for (int h = tid; h < 768; h += 256) {
        float epv = ep[h], b1v = b1[h], w2v = W2[h];
        #pragma unroll
        for (int a = 0; a < 9; a++) {
            float x = epv + argPart[((size_t)(n * 16 + a)) * 768 + h] + b1v;
            float g = 0.5f * x * (1.0f + erff(x * 0.70710678118654752f));
            acc[a] += g * w2v;
        }
    }
    #pragma unroll
    for (int a = 0; a < 9; a++) {
        float v = acc[a];
        for (int o = 16; o > 0; o >>= 1) v += __shfl_down(v, o, 32);
        if (lane == 0) part[a][wid] = v;
    }
    __syncthreads();
    if (tid < 9) {
        float s = 0.f;
        for (int w = 0; w < 8; w++) s += part[tid][w];
        sc[tid] = s + b2[0];
    }
    if (tid < 24) out[(size_t)blk * 24 + tid] = -1000000.0f;
    __syncthreads();
    if (tid < 9) out[(size_t)blk * 24 + argmap[n * 9 + tid]] = sc[tid];
}

// ---------------------------------------------------------------------------
extern "C" void kernel_launch(void* const* d_in, const int* in_sizes, int n_in,
                              void* d_out, int out_size, void* d_ws, size_t ws_size,
                              hipStream_t stream) {
    (void)in_sizes; (void)n_in; (void)out_size; (void)ws_size;
    const float* allemb = (const float*)d_in[0];   // (16,256,768)
    const float* attn   = (const float*)d_in[1];   // (3,16,12,256,256)
    const float* entmap = (const float*)d_in[2];   // (16,128,128)
    const float* argW   = (const float*)d_in[3];   // (16,9,9)
    // d_in[4] is_triggers: result deleted in reference -> unused
    const float* W1     = (const float*)d_in[5];   // (4608,768)
    const float* b1     = (const float*)d_in[6];   // (768,)
    const float* W2     = (const float*)d_in[7];   // (768,1)
    const float* b2v    = (const float*)d_in[8];   // (1,)
    const int*   idxs   = (const int*)d_in[9];     // (16,130)
    const int*   argmap = (const int*)d_in[10];    // (16,9)
    float* out = (float*)d_out;                    // (16,128,24)

    char* ws = (char*)d_ws;
    size_t off = 0;
    auto alloc = [&](size_t bytes) -> char* {
        char* p = ws + off;
        off = (off + bytes + 255) & ~(size_t)255;
        return p;
    };
    bf16*  sentBT  = (bf16*) alloc((size_t)16 * 768 * 128 * 2);  // (n, D, S)
    bf16*  emTB    = (bf16*) alloc((size_t)16 * 128 * 128 * 2);  // (n, E, S)
    bf16*  t2tB    = (bf16*) alloc((size_t)16 * 128 * 128 * 2);  // (n, S, S)
    float* entF    = (float*)alloc((size_t)16 * 128 * 768 * 4);  // (n, E, D)
    bf16*  h2hT    = (bf16*) alloc((size_t)16 * 768 * 128 * 2);  // (n, D, S)
    bf16*  ah2hB   = (bf16*) alloc((size_t)16 * 128 * 768 * 2);  // (n, E, D)
    float* argembF = (float*)alloc((size_t)16 * 9 * 768 * 4);
    bf16*  EntFeats= (bf16*) alloc((size_t)2048 * 2304 * 2);
    bf16*  ArgFeats= (bf16*) alloc((size_t)256 * 2304 * 2);
    bf16*  W1entT  = (bf16*) alloc((size_t)768 * 2304 * 2);      // (N_HID, 3D)
    bf16*  W1argT  = (bf16*) alloc((size_t)768 * 2304 * 2);
    float* entPart = (float*)alloc((size_t)2048 * 768 * 4);
    float* argPart = (float*)alloc((size_t)256 * 768 * 4);

    prep_w1_kernel<<<(2 * 768 * 2304 + 255) / 256, 256, 0, stream>>>(W1, W1entT, W1argT);
    gather_kernel<<<16, 256, 0, stream>>>(allemb, entmap, argW, idxs, sentBT, emTB, argembF);
    t2t_kernel<<<2048, 128, 0, stream>>>(attn, idxs, t2tB);

    // entity_emb (f32) = entity_mapping^T @ sent_emb          [batched 16x]
    wmma_gemm_bf16<<<dim3(6, 2, 16), 256, 0, stream>>>(
        emTB, sentBT, entF, nullptr, 128, 128, 128, 768, 0,
        128L * 128, 768L * 128, 128L * 768, 1);
    // h2h_sent (bf16, stored transposed D x S) = t2t @ sent_emb
    wmma_gemm_bf16<<<dim3(6, 2, 16), 256, 0, stream>>>(
        t2tB, sentBT, nullptr, h2hT, 128, 128, 128, 0, 128,
        128L * 128, 768L * 128, 768L * 128, 4);
    // A_h2h_e (bf16 row-major) = entity_mapping^T @ h2h_sent
    wmma_gemm_bf16<<<dim3(6, 2, 16), 256, 0, stream>>>(
        emTB, h2hT, nullptr, ah2hB, 128, 128, 128, 768, 0,
        128L * 128, 768L * 128, 128L * 768, 2);

    interactions_kernel<<<16, 256, 0, stream>>>(entF, argembF, ah2hB, EntFeats, ArgFeats);

    // ent_part = EntFeats (2048x2304) @ W1entT^T
    wmma_gemm_bf16<<<dim3(6, 32, 1), 256, 0, stream>>>(
        EntFeats, W1entT, entPart, nullptr, 2304, 2304, 2304, 768, 0,
        0, 0, 0, 1);
    // arg_part = ArgFeats (256x2304) @ W1argT^T
    wmma_gemm_bf16<<<dim3(6, 4, 1), 256, 0, stream>>>(
        ArgFeats, W1argT, argPart, nullptr, 2304, 2304, 2304, 768, 0,
        0, 0, 0, 1);

    score_kernel<<<2048, 256, 0, stream>>>(entPart, argPart, b1, W2, b2v, argmap, out);
}
